// NADE_10316511445032
// MI455X (gfx1250) — matrix operand order repaired and checked by reading, not compiled, not run on hardware
//
#include <hip/hip_runtime.h>
#include <hip/hip_bf16.h>
#include <math.h>

// ---------------------------------------------------------------------------
// NADE block for MI455X (gfx1250, wave32, WMMA).
// One batched-strided WMMA f16->f32 GEMM (2x2 register tiling, contiguous-B
// specialization) carries DCT/IDCT, QKV, QK^T, PV, proj, and the 1x1 refine.
// ---------------------------------------------------------------------------

typedef __attribute__((ext_vector_type(16))) _Float16 v16h;
typedef __attribute__((ext_vector_type(8)))  float    v8f;

#define PI_F 3.14159265358979323846f

__device__ __forceinline__ float sigmoidf_(float v) { return 1.0f / (1.0f + expf(-v)); }
__device__ __forceinline__ int   imin_(int a, int b) { return a < b ? a : b; }

// ---------------- generic batched WMMA GEMM (2x2 tiles / wave) -------------
// C[bi] = alpha * A[bi](MxK, row stride lda) @ B[bi](KxN, strides ldbK/ldbN) + bias
// Output strides ldcM/ldcN allow transposed stores (used to keep every DCT
// chain GEMM on the contiguous-B fast path).
// Batch offset per tensor: (bi/bdiv)*s1 + (bi%bdiv)*s2.
// K must be a multiple of 32. M,N arbitrary (clamped loads, guarded stores).
template <bool BK1>
__global__ __launch_bounds__(128) void gemm_wmma_f16(
    const _Float16* __restrict__ A,  long sA1, long sA2, int lda,
    const _Float16* __restrict__ Bm, long sB1, long sB2, int ldbK, int ldbN,
    float* __restrict__ Cf, _Float16* __restrict__ Ch, long sC1, long sC2,
    int ldcM, int ldcN,
    const float* __restrict__ bias, float alpha, int M, int N, int K, int bdiv)
{
  const int lane = threadIdx.x;                 // 0..31 (wave32)
  const int wv   = threadIdx.y;                 // 0..3
  const int tnb  = (blockIdx.x * 4 + wv) * 32;  // N block (32 wide)
  const int tmb  = blockIdx.y * 32;             // M block (32 tall)
  const int bi   = blockIdx.z;
  if (tnb >= N) return;

  const int hi  = lane >> 4;                    // lane half
  const int l16 = lane & 15;
  const long oA = (long)(bi / bdiv) * sA1 + (long)(bi % bdiv) * sA2;
  const long oB = (long)(bi / bdiv) * sB1 + (long)(bi % bdiv) * sB2;
  const long oC = (long)(bi / bdiv) * sC1 + (long)(bi % bdiv) * sC2;

  // clamped fragment rows/cols (results for clamped lanes discarded at store)
  const _Float16* Ap0 = A + oA + (long)imin_(tmb + l16,      M - 1) * lda;
  const _Float16* Ap1 = A + oA + (long)imin_(tmb + 16 + l16, M - 1) * lda;
  const _Float16* Bq0 = Bm + oB + (long)imin_(tnb + l16,      N - 1) * ldbN;
  const _Float16* Bq1 = Bm + oB + (long)imin_(tnb + 16 + l16, N - 1) * ldbN;
  const long bstep = BK1 ? 1 : (long)ldbK;

  v8f acc00, acc01, acc10, acc11;
#pragma unroll
  for (int r = 0; r < 8; ++r) { acc00[r] = 0.0f; acc01[r] = 0.0f; acc10[r] = 0.0f; acc11[r] = 0.0f; }

  for (int kb = 0; kb < K; kb += 32) {
    v16h a0, a1, b0, b1;
    // A fragment (16x32): M = lane&15, K = (lane>=16?16:0)+e
    const int ka0 = kb + hi * 16;
#pragma unroll
    for (int e = 0; e < 8; ++e) { a0[e] = Ap0[ka0 + e];      a1[e] = Ap1[ka0 + e]; }
#pragma unroll
    for (int e = 0; e < 8; ++e) { a0[8 + e] = Ap0[ka0 + 8 + e]; a1[8 + e] = Ap1[ka0 + 8 + e]; }
    // B fragment (32x16): N = lane&15, K = (lane>=16?16:0)+e
    const _Float16* Bp0 = Bq0 + (long)(kb + hi * 16) * bstep;
    const _Float16* Bp1 = Bq1 + (long)(kb + hi * 16) * bstep;
#pragma unroll
    for (int e = 0; e < 16; ++e) { b0[e] = Bp0[e * bstep]; b1[e] = Bp1[e * bstep]; }
    if (kb + 32 < K) {
      __builtin_prefetch(Ap0 + ka0 + 32, 0, 1);   // global_prefetch_b8
      __builtin_prefetch(Ap1 + ka0 + 32, 0, 1);
    }
    acc00 = __builtin_amdgcn_wmma_f32_16x16x32_f16(false, a0, false, b0, (short)0, acc00, false, false);
    acc01 = __builtin_amdgcn_wmma_f32_16x16x32_f16(false, a0, false, b1, (short)0, acc01, false, false);
    acc10 = __builtin_amdgcn_wmma_f32_16x16x32_f16(false, a1, false, b0, (short)0, acc10, false, false);
    acc11 = __builtin_amdgcn_wmma_f32_16x16x32_f16(false, a1, false, b1, (short)0, acc11, false, false);
  }

  // C/D layout: lane l16 = N col; VGPR r -> M = r + hi*8
#pragma unroll
  for (int sj = 0; sj < 2; ++sj) {
    const int n = tnb + sj * 16 + l16;
    if (n >= N) continue;
    const float bv = bias ? bias[n] : 0.0f;
#pragma unroll
    for (int si = 0; si < 2; ++si) {
      const v8f* ap = (si == 0) ? ((sj == 0) ? &acc00 : &acc01)
                                : ((sj == 0) ? &acc10 : &acc11);
#pragma unroll
      for (int r = 0; r < 8; ++r) {
        const int m = tmb + si * 16 + r + hi * 8;
        if (m >= M) continue;
        const float v  = (*ap)[r] * alpha + bv;
        const long idx = oC + (long)m * ldcM + (long)n * ldcN;
        if (Cf) Cf[idx] = v;
        if (Ch) Ch[idx] = (_Float16)v;
      }
    }
  }
}

// ---------------- init / convert kernels ----------------------------------
__global__ void k_init_basis(_Float16* Bh, _Float16* iBh) {
  int idx = blockIdx.x * 256 + threadIdx.x;            // 96*96
  if (idx >= 9216) return;
  int n = idx % 96, k = idx / 96;
  float c = cosf(PI_F * (float)k * (float)(2 * n + 1) / 192.0f);
  Bh[k * 96 + n] = (_Float16)(2.0f * c);               // forward DCT-II (factor 2)
  float ic = ((k == 0) ? 0.5f : c) * (1.0f / 192.0f);  // inverse basis
  iBh[n * 96 + k] = (_Float16)ic;
}

__global__ void k_init_bias(const float* __restrict__ rpb, float* __restrict__ biasT) {
  int idx = blockIdx.x * 256 + threadIdx.x;            // 8*144*144
  if (idx >= 8 * 144 * 144) return;
  int m = idx % 144, l = (idx / 144) % 144, h = idx / (144 * 144);
  int i1 = l / 12, j1 = l % 12, i2 = m / 12, j2 = m % 12;
  int rel = (i1 - i2 + 11) * 23 + (j1 - j2 + 11);
  biasT[idx] = rpb[rel * 8 + h];
}

__global__ void k_f32_to_f16(const float* __restrict__ in, _Float16* __restrict__ out, long n) {
  long i = (long)blockIdx.x * 256 + threadIdx.x;
  if (i < n) out[i] = (_Float16)in[i];
}

__global__ void k_zero_h(_Float16* p, long n) {
  long i = (long)blockIdx.x * 256 + threadIdx.x;
  if (i < n) p[i] = (_Float16)0.0f;
}

// ---------------- FreqNoiseDetector ---------------------------------------
__global__ __launch_bounds__(256) void k_bc_minmax(const float* __restrict__ F, float* __restrict__ mnmx) {
  int bc = blockIdx.x, t = threadIdx.x;
  float mn = 1e30f, mx = -1e30f;
  for (int i = t; i < 9216; i += 256) {
    float v = fabsf(F[(long)bc * 9216 + i]);
    mn = fminf(mn, v); mx = fmaxf(mx, v);
  }
  __shared__ float smn[256], smx[256];
  smn[t] = mn; smx[t] = mx; __syncthreads();
  for (int s = 128; s > 0; s >>= 1) {
    if (t < s) { smn[t] = fminf(smn[t], smn[t + s]); smx[t] = fmaxf(smx[t], smx[t + s]); }
    __syncthreads();
  }
  if (t == 0) { mnmx[bc * 2] = smn[0]; mnmx[bc * 2 + 1] = smx[0]; }
}

__global__ __launch_bounds__(256) void k_chan_mean(const float* __restrict__ F,
                                                   const float* __restrict__ mnmx,
                                                   float* __restrict__ fmean) {
  int idx = blockIdx.x * 256 + threadIdx.x;            // 8*9216
  if (idx >= 73728) return;
  int b = idx / 9216, hw = idx % 9216;
  float s = 0.0f;
  for (int c = 0; c < 256; ++c) {
    int bc = b * 256 + c;
    float v = fabsf(F[(long)bc * 9216 + hw]);
    s += (v - mnmx[bc * 2]) / (mnmx[bc * 2 + 1] + 1e-6f);
  }
  fmean[idx] = s * (1.0f / 256.0f);
}

__global__ __launch_bounds__(256) void k_conv3x3(const float* __restrict__ in,
                                                 const float* __restrict__ w,
                                                 float* __restrict__ out) {
  int idx = blockIdx.x * 256 + threadIdx.x;            // 8*9216
  if (idx >= 73728) return;
  int wd = idx % 96, h = (idx / 96) % 96, b = idx / 9216;
  float s = 0.0f;
  for (int ky = 0; ky < 3; ++ky) {
    int yy = h + ky - 1; if (yy < 0 || yy >= 96) continue;
    for (int kx = 0; kx < 3; ++kx) {
      int xx = wd + kx - 1; if (xx < 0 || xx >= 96) continue;
      s += in[b * 9216 + yy * 96 + xx] * w[ky * 3 + kx];
    }
  }
  out[idx] = s;
}

__global__ __launch_bounds__(256) void k_bn1_stats(const float* __restrict__ E, float* __restrict__ red) {
  int t = threadIdx.x;
  float s = 0.0f, s2 = 0.0f;
  for (int i = t; i < 73728; i += 256) { float v = E[i]; s += v; s2 += v * v; }
  __shared__ float a1[256], a2[256];
  a1[t] = s; a2[t] = s2; __syncthreads();
  for (int st = 128; st > 0; st >>= 1) {
    if (t < st) { a1[t] += a1[t + st]; a2[t] += a2[t + st]; }
    __syncthreads();
  }
  if (t == 0) {
    float m = a1[0] * (1.0f / 73728.0f);
    red[0] = m;
    red[1] = a2[0] * (1.0f / 73728.0f) - m * m;
  }
  if (t >= 2 && t < 18) red[t] = 0.0f;                 // zero pooled accumulators (determinism)
}

__global__ __launch_bounds__(256) void k_energy(const float* __restrict__ E, float* __restrict__ red,
                                                const float* __restrict__ g, const float* __restrict__ bparm,
                                                float* __restrict__ le) {
  int idx = blockIdx.x * 256 + threadIdx.x;            // 73728, 36 blocks per batch
  int b = idx / 9216;
  float e = (E[idx] - red[0]) * rsqrtf(red[1] + 1e-5f) * g[0] + bparm[0];
  e = fmaxf(e, 0.0f);
  le[idx] = e;
  __shared__ float s[256];
  s[threadIdx.x] = e; __syncthreads();
  for (int st = 128; st > 0; st >>= 1) {
    if (threadIdx.x < st) s[threadIdx.x] += s[threadIdx.x + st];
    __syncthreads();
  }
  if (threadIdx.x == 0) atomicAdd(&red[2 + b], s[0]);
}

__global__ void k_noiseg(float* __restrict__ red, const float* __restrict__ nlw, const float* __restrict__ nlb) {
  int b = threadIdx.x;                                 // 8
  if (b < 8) red[10 + b] = sigmoidf_((red[2 + b] * (1.0f / 9216.0f)) * nlw[0] + nlb[0]);
}

__global__ __launch_bounds__(256) void k_mask(const float* __restrict__ le, const float* __restrict__ red,
                                              float* __restrict__ nm) {
  int idx = blockIdx.x * 256 + threadIdx.x;            // 73728
  if (idx >= 73728) return;
  nm[idx] = sigmoidf_(1.0f - le[idx]) * red[10 + idx / 9216];
}

// ---------------- attention data movement ----------------------------------
__global__ __launch_bounds__(256) void k_pack_tokens(const float* __restrict__ x, _Float16* __restrict__ T) {
  long idx = (long)blockIdx.x * 256 + threadIdx.x;     // 73728*256
  int c = (int)(idx & 255);
  long r = idx >> 8;
  int l = (int)(r % 144); int wxy = (int)(r / 144);
  int wx = wxy & 7, wy = (wxy >> 3) & 7, b = wxy >> 6;
  int h = wy * 12 + l / 12, wd = wx * 12 + l % 12;
  T[idx] = (_Float16)x[(((long)b * 256 + c) * 96 + h) * 96 + wd];
}

__global__ __launch_bounds__(256) void k_softmax(const float* __restrict__ S,
                                                 const float* __restrict__ biasT,
                                                 _Float16* __restrict__ P) {
  int row = blockIdx.x;                                // 4096*144
  int bIdx = row / 144, l = row % 144, head = bIdx & 7;
  int t = threadIdx.x;
  __shared__ float red[256];
  float v = -1e30f;
  if (t < 144) v = S[(long)row * 144 + t] + biasT[((head * 144) + l) * 144 + t];
  red[t] = v; __syncthreads();
  for (int s = 128; s > 0; s >>= 1) { if (t < s) red[t] = fmaxf(red[t], red[t + s]); __syncthreads(); }
  float mx = red[0]; __syncthreads();
  float e = (t < 144) ? expf(v - mx) : 0.0f;
  red[t] = e; __syncthreads();
  for (int s = 128; s > 0; s >>= 1) { if (t < s) red[t] += red[t + s]; __syncthreads(); }
  float inv = 1.0f / red[0];
  if (t < 160) P[(long)row * 160 + t] = (_Float16)((t < 144) ? e * inv : 0.0f);  // zero-pad K->160
}

__global__ __launch_bounds__(256) void k_unwindow_lpe(const float* __restrict__ proj,
                                                      const float* __restrict__ x,
                                                      const float* __restrict__ lpw,
                                                      const float* __restrict__ lpb,
                                                      const float* __restrict__ lps,
                                                      const float* __restrict__ nm,
                                                      _Float16* __restrict__ xa_h) {
  long idx = (long)blockIdx.x * 256 + threadIdx.x;     // NCHW 8*256*96*96
  int wd = (int)(idx % 96), h = (int)((idx / 96) % 96);
  int c = (int)((idx / 9216) % 256), b = (int)(idx / (9216 * 256));
  long r = (((long)b * 8 + h / 12) * 8 + wd / 12) * 144 + (h % 12) * 12 + (wd % 12);
  float attn = proj[r * 256 + c];
  float s = 0.0f;
  for (int ky = 0; ky < 3; ++ky) {
    int yy = h + ky - 1; if (yy < 0 || yy >= 96) continue;
    for (int kx = 0; kx < 3; ++kx) {
      int xx = wd + kx - 1; if (xx < 0 || xx >= 96) continue;
      s += x[(((long)b * 256 + c) * 96 + yy) * 96 + xx] * lpw[c * 9 + ky * 3 + kx];
    }
  }
  float lpe = (s + lpb[c]) * lps[0] * nm[(long)b * 9216 + h * 96 + wd];
  xa_h[idx] = (_Float16)(attn + lpe);
}

// ---------------- frequency gate / final -----------------------------------
__global__ __launch_bounds__(256) void k_bc_mean(const float* __restrict__ F, float* __restrict__ fm) {
  int bc = blockIdx.x, t = threadIdx.x;
  float s = 0.0f;
  for (int i = t; i < 9216; i += 256) s += F[(long)bc * 9216 + i];
  __shared__ float a[256];
  a[t] = s; __syncthreads();
  for (int st = 128; st > 0; st >>= 1) { if (t < st) a[t] += a[t + st]; __syncthreads(); }
  if (t == 0) fm[bc] = a[0] * (1.0f / 9216.0f);
}

// writes the gated spectrum TRANSPOSED per (b,c): fgT[kw][kh], so the IDCT-H
// GEMM sees a contiguous-K B operand.
__global__ __launch_bounds__(256) void k_gate(const float* __restrict__ F, const float* __restrict__ fm,
                                              _Float16* __restrict__ outT) {
  long idx = (long)blockIdx.x * 256 + threadIdx.x;
  int bc = (int)(idx / 9216), rem = (int)(idx % 9216);
  int kh = rem / 96, kw = rem % 96;
  float f = F[idx];
  float g = sigmoidf_(5.0f * (fabsf(f) - fm[bc]));
  outT[(long)bc * 9216 + kw * 96 + kh] = (_Float16)(f * g);
}

__global__ __launch_bounds__(256) void k_pack_pix(const float* __restrict__ x, const float* __restrict__ xf,
                                                  _Float16* __restrict__ T) {
  long idx = (long)blockIdx.x * 256 + threadIdx.x;     // 73728*256 tokens = (b,h,w) x c
  int c = (int)(idx & 255);
  long p = idx >> 8;
  int b = (int)(p / 9216), hw = (int)(p % 9216);
  long src = ((long)b * 256 + c) * 9216 + hw;
  T[idx] = (_Float16)(x[src] + xf[src]);
}

__global__ __launch_bounds__(256) void k_bn2_stats(const float* __restrict__ R, float* __restrict__ cs) {
  int c = blockIdx.x, t = threadIdx.x;
  float s = 0.0f, s2 = 0.0f;
  for (long r = t; r < 73728; r += 256) { float v = R[r * 256 + c]; s += v; s2 += v * v; }
  __shared__ float a1[256], a2[256];
  a1[t] = s; a2[t] = s2; __syncthreads();
  for (int st = 128; st > 0; st >>= 1) {
    if (t < st) { a1[t] += a1[t + st]; a2[t] += a2[t + st]; }
    __syncthreads();
  }
  if (t == 0) {
    float m = a1[0] * (1.0f / 73728.0f);
    cs[c * 2] = m;
    cs[c * 2 + 1] = a2[0] * (1.0f / 73728.0f) - m * m;
  }
}

__global__ __launch_bounds__(256) void k_final(const float* __restrict__ R, const float* __restrict__ cs,
                                               const float* __restrict__ g, const float* __restrict__ bb,
                                               float* __restrict__ out) {
  long idx = (long)blockIdx.x * 256 + threadIdx.x;     // NCHW
  int hw = (int)(idx % 9216), c = (int)((idx / 9216) % 256), b = (int)(idx / (9216 * 256));
  long r = (long)b * 9216 + hw;
  float y = (R[r * 256 + c] - cs[c * 2]) * rsqrtf(cs[c * 2 + 1] + 1e-5f) * g[c] + bb[c];
  out[idx] = 0.5f * y * (1.0f + erff(y * 0.70710678118f));
}

// ---------------------------------------------------------------------------
static inline void launch_gemm(hipStream_t s,
    const _Float16* A, long sA1, long sA2, int lda,
    const _Float16* Bm, long sB1, long sB2, int ldbK, int ldbN,
    float* Cf, _Float16* Ch, long sC1, long sC2, int ldcM, int ldcN,
    const float* bias, float alpha, int M, int N, int K, int batch, int bdiv) {
  dim3 g((N + 127) / 128, (M + 31) / 32, batch), b(32, 4);
  if (ldbK == 1)
    gemm_wmma_f16<true><<<g, b, 0, s>>>(A, sA1, sA2, lda, Bm, sB1, sB2, ldbK, ldbN,
                                        Cf, Ch, sC1, sC2, ldcM, ldcN, bias, alpha, M, N, K, bdiv);
  else
    gemm_wmma_f16<false><<<g, b, 0, s>>>(A, sA1, sA2, lda, Bm, sB1, sB2, ldbK, ldbN,
                                         Cf, Ch, sC1, sC2, ldcM, ldcN, bias, alpha, M, N, K, bdiv);
}

extern "C" void kernel_launch(void* const* d_in, const int* in_sizes, int n_in,
                              void* d_out, int out_size, void* d_ws, size_t ws_size,
                              hipStream_t stream) {
  (void)in_sizes; (void)n_in; (void)out_size; (void)ws_size;
  const float* x      = (const float*)d_in[0];
  const float* qkv_w  = (const float*)d_in[1];
  const float* qkv_b  = (const float*)d_in[2];
  const float* proj_w = (const float*)d_in[3];
  const float* proj_b = (const float*)d_in[4];
  const float* lpe_w  = (const float*)d_in[5];
  const float* lpe_b  = (const float*)d_in[6];
  const float* lpe_s  = (const float*)d_in[7];
  const float* rpb    = (const float*)d_in[8];
  const float* ec_w   = (const float*)d_in[9];
  const float* bn1_g  = (const float*)d_in[10];
  const float* bn1_b  = (const float*)d_in[11];
  const float* nl_w   = (const float*)d_in[12];
  const float* nl_b   = (const float*)d_in[13];
  const float* fr_w   = (const float*)d_in[14];
  const float* bn2_g  = (const float*)d_in[15];
  const float* bn2_b  = (const float*)d_in[16];

  // ---- workspace bump allocator (256B aligned) ----
  char* base = (char*)d_ws;
  size_t off = 0;
  auto alloc = [&](size_t bytes) -> void* {
    void* p = base + off;
    off = (off + bytes + 255) & ~(size_t)255;
    return p;
  };
  const size_t IMG = 18874368;                          // 8*256*96*96
  _Float16* Basis  = (_Float16*)alloc(9216 * 2);
  _Float16* iBasis = (_Float16*)alloc(9216 * 2);
  _Float16* wq     = (_Float16*)alloc(196608 * 2);      // qkv_w f16 [768][256]
  _Float16* wp     = (_Float16*)alloc(65536 * 2);       // proj_w f16
  _Float16* wf     = (_Float16*)alloc(65536 * 2);       // fr_w f16
  float* biasT  = (float*)alloc(165888 * 4);            // rel-pos bias [8][144][144]
  float* mnmx   = (float*)alloc(4096 * 4);
  float* fmean  = (float*)alloc(73728 * 4);
  float* convE  = (float*)alloc(73728 * 4);
  float* red    = (float*)alloc(64 * 4);
  float* nm     = (float*)alloc(73728 * 4);
  float* fmean2 = (float*)alloc(2048 * 4);
  float* cs     = (float*)alloc(512 * 4);
  _Float16* BUFA = (_Float16*)alloc(IMG * 2);           // x_h / tokens / xa_h / fgT / pixtok
  _Float16* BUFB = (_Float16*)alloc(IMG * 2);           // DCT intermediates (transposed)
  float*    BUFF = (float*)alloc(IMG * 4);              // freq / proj-out / freq2 / xfreq / refine-out
  _Float16* QKVh = (_Float16*)alloc(((size_t)73728 + 16) * 768 * 2);
  float*    S    = (float*)alloc((size_t)4096 * 20736 * 4);
  _Float16* Ph   = (_Float16*)alloc((size_t)4096 * 23040 * 2);

  const dim3 B256(256);
  // ---- init & converts ----
  k_init_basis<<<dim3(36), B256, 0, stream>>>(Basis, iBasis);
  k_init_bias<<<dim3((165888 + 255) / 256), B256, 0, stream>>>(rpb, biasT);
  k_f32_to_f16<<<dim3(73728), B256, 0, stream>>>(x, BUFA, (long)IMG);
  k_f32_to_f16<<<dim3(768), B256, 0, stream>>>(qkv_w, wq, 196608);
  k_f32_to_f16<<<dim3(256), B256, 0, stream>>>(proj_w, wp, 65536);
  k_f32_to_f16<<<dim3(256), B256, 0, stream>>>(fr_w, wf, 65536);
  k_zero_h<<<dim3(48), B256, 0, stream>>>(QKVh + (size_t)73728 * 768, 12288);  // PV K-pad rows

  // ---- dct2(x): T1t[kw][h] = (X @ Basis^T)^T, then freq = Basis @ T1 ----
  launch_gemm(stream, BUFA, 9216, 0, 96, Basis, 0, 0, 1, 96,
              nullptr, BUFB, 9216, 0, /*ldcM*/1, /*ldcN*/96, nullptr, 1.0f, 96, 96, 96, 2048, 1);
  launch_gemm(stream, Basis, 0, 0, 96, BUFB, 9216, 0, 1, 96,
              BUFF, nullptr, 9216, 0, 96, 1, nullptr, 1.0f, 96, 96, 96, 2048, 1);

  // ---- FreqNoiseDetector ----
  k_bc_minmax<<<dim3(2048), B256, 0, stream>>>(BUFF, mnmx);
  k_chan_mean<<<dim3(288), B256, 0, stream>>>(BUFF, mnmx, fmean);
  k_conv3x3<<<dim3(288), B256, 0, stream>>>(fmean, ec_w, convE);
  k_bn1_stats<<<dim3(1), B256, 0, stream>>>(convE, red);
  k_energy<<<dim3(288), B256, 0, stream>>>(convE, red, bn1_g, bn1_b, convE);
  k_noiseg<<<dim3(1), dim3(8), 0, stream>>>(red, nl_w, nl_b);
  k_mask<<<dim3(288), B256, 0, stream>>>(convE, red, nm);

  // ---- attention ----
  k_pack_tokens<<<dim3(73728), B256, 0, stream>>>(x, BUFA);                   // tokens f16
  launch_gemm(stream, BUFA, 0, 0, 256, wq, 0, 0, 1, 256,                      // QKV = T @ Wqkv^T + b
              nullptr, QKVh, 0, 0, 768, 1, qkv_b, 1.0f, 73728, 768, 256, 1, 1);
  launch_gemm(stream, QKVh, 110592, 32, 768,                                  // S = scale * Q @ K^T
              QKVh + 256, 110592, 32, 1, 768,
              S, nullptr, 165888, 20736, 144, 1, nullptr, 0.17677669529663687f,
              144, 144, 32, 4096, 8);
  k_softmax<<<dim3(589824), B256, 0, stream>>>(S, biasT, Ph);                 // +bias, softmax, pad->160
  launch_gemm(stream, Ph, 184320, 23040, 160,                                 // O = P @ V (heads merged)
              QKVh + 512, 110592, 32, 768, 1,
              nullptr, BUFA, 36864, 32, 256, 1, nullptr, 1.0f, 144, 32, 160, 4096, 8);
  launch_gemm(stream, BUFA, 0, 0, 256, wp, 0, 0, 1, 256,                      // proj
              BUFF, nullptr, 0, 0, 256, 1, proj_b, 1.0f, 73728, 256, 256, 1, 1);
  k_unwindow_lpe<<<dim3(73728), B256, 0, stream>>>(BUFF, x, lpe_w, lpe_b, lpe_s, nm, BUFA);

  // ---- frequency soft gate: dct2 -> gate -> idct2 ----
  launch_gemm(stream, BUFA, 9216, 0, 96, Basis, 0, 0, 1, 96,                  // T1t (transposed store)
              nullptr, BUFB, 9216, 0, 1, 96, nullptr, 1.0f, 96, 96, 96, 2048, 1);
  launch_gemm(stream, Basis, 0, 0, 96, BUFB, 9216, 0, 1, 96,                  // freq2
              BUFF, nullptr, 9216, 0, 96, 1, nullptr, 1.0f, 96, 96, 96, 2048, 1);
  k_bc_mean<<<dim3(2048), B256, 0, stream>>>(BUFF, fmean2);
  k_gate<<<dim3(73728), B256, 0, stream>>>(BUFF, fmean2, BUFA);               // fgT (transposed)
  launch_gemm(stream, iBasis, 0, 0, 96, BUFA, 9216, 0, 1, 96,                 // T2 = iB @ fg
              nullptr, BUFB, 9216, 0, 96, 1, nullptr, 1.0f, 96, 96, 96, 2048, 1);
  launch_gemm(stream, BUFB, 9216, 0, 96, iBasis, 0, 0, 1, 96,                 // xfreq = T2 @ iB^T
              BUFF, nullptr, 9216, 0, 96, 1, nullptr, 1.0f, 96, 96, 96, 2048, 1);

  // ---- 1x1 refine + BN + GELU ----
  k_pack_pix<<<dim3(73728), B256, 0, stream>>>(x, BUFF, BUFA);                // x + x_freq -> f16 tokens
  launch_gemm(stream, BUFA, 0, 0, 256, wf, 0, 0, 1, 256,
              BUFF, nullptr, 0, 0, 256, 1, nullptr, 1.0f, 73728, 256, 256, 1, 1);
  k_bn2_stats<<<dim3(256), B256, 0, stream>>>(BUFF, cs);
  k_final<<<dim3(73728), B256, 0, stream>>>(BUFF, cs, bn2_g, bn2_b, (float*)d_out);
}
